// HierarchicalBPModel_58884001628747
// MI455X (gfx1250) — compile-verified, matrix operand order checked by reading, not run on hardware
//
#include <hip/hip_runtime.h>
#include <cmath>

#define NPATH  1024
#define NTOK   64
#define DMODEL 512
#define NHEAD  8
#define HDIM   64
#define NSAMP  64
#define NCAT   30
#define NCLS   1085
#define FFDIM  2048

// padded LDS leading dims (elements) to avoid 64-bank conflicts
#define LDH  516   // f32 residual stream
#define LDX  520   // f16 [64,512] buffers
#define LDQ  72    // f16 [64,64] buffers
#define LDSC 68    // f32 [64,64] scores

typedef __attribute__((ext_vector_type(16))) _Float16 v16h;
typedef __attribute__((ext_vector_type(8)))  float    v8f;

// ---------------------------------------------------------------------------
// WMMA fragment helpers (CDNA5 ISA 7.12.2 layouts, wave32)
// ---------------------------------------------------------------------------
__device__ __forceinline__ int kofs(int j, int half) {
  // 16-bit A-matrix 16x32: vec elem j<8 -> K=j+8*half ; j>=8 -> K=j+8+8*half
  return (j < 8 ? j : j + 8) + (half << 3);
}
__device__ __forceinline__ v8f zero8() {
  v8f c;
#pragma unroll
  for (int r = 0; r < 8; ++r) c[r] = 0.f;
  return c;
}
// A tile 16x32 from row-major f16 source, leading dim ld, at (r0, k0)
__device__ __forceinline__ v16h ldA(const _Float16* S, int ld, int r0, int k0, int lane) {
  int m = r0 + (lane & 15); int h = lane >> 4; v16h a;
#pragma unroll
  for (int j = 0; j < 16; ++j) a[j] = S[(size_t)m * ld + k0 + kofs(j, h)];
  return a;
}
// B tile 32x16 where B[e][n] = S[(n0+n)*ld + k0+e]   (X @ W^T style: S holds W)
__device__ __forceinline__ v16h ldBT(const _Float16* S, int ld, int k0, int n0, int lane) {
  int n = n0 + (lane & 15); int h = lane >> 4; v16h b;
#pragma unroll
  for (int j = 0; j < 16; ++j) b[j] = S[(size_t)n * ld + k0 + kofs(j, h)];
  return b;
}
__device__ __forceinline__ v8f ldC(const float* S, int ld, int r0, int n0, int lane) {
  int n = n0 + (lane & 15); int h = lane >> 4; v8f c;
#pragma unroll
  for (int r = 0; r < 8; ++r) c[r] = S[(size_t)(r0 + r + (h << 3)) * ld + n];
  return c;
}
__device__ __forceinline__ void stC(float* S, int ld, int r0, int n0, int lane, v8f c) {
  int n = n0 + (lane & 15); int h = lane >> 4;
#pragma unroll
  for (int r = 0; r < 8; ++r) S[(size_t)(r0 + r + (h << 3)) * ld + n] = c[r];
}
__device__ __forceinline__ void stC16(_Float16* S, int ld, int r0, int n0, int lane,
                                      v8f c, bool dorelu) {
  int n = n0 + (lane & 15); int h = lane >> 4;
#pragma unroll
  for (int r = 0; r < 8; ++r) {
    float x = c[r]; if (dorelu) x = fmaxf(x, 0.f);
    S[(size_t)(r0 + r + (h << 3)) * ld + n] = (_Float16)x;
  }
}
// transposed store: tile element (m, n) -> S[n*ld + m]   (contiguous per lane)
__device__ __forceinline__ void stC16T(_Float16* S, int ld, int r0, int n0, int lane,
                                       v8f c) {
  int n = n0 + (lane & 15); int h = lane >> 4;
#pragma unroll
  for (int r = 0; r < 8; ++r)
    S[(size_t)n * ld + r0 + r + (h << 3)] = (_Float16)c[r];
}
__device__ __forceinline__ v8f wmma16(v16h a, v16h b, v8f c) {
  return __builtin_amdgcn_wmma_f32_16x16x32_f16(false, a, false, b, (short)0, c,
                                                false, false);
}

// ---------------------------------------------------------------------------
// weight f32 -> f16 conversion
// ---------------------------------------------------------------------------
__global__ void f32_to_f16_kernel(const float* __restrict__ s, _Float16* __restrict__ d,
                                  long n) {
  long i = (long)blockIdx.x * blockDim.x + threadIdx.x;
  long stride = (long)gridDim.x * blockDim.x;
  for (; i < n; i += stride) d[i] = (_Float16)s[i];
}

// ---------------------------------------------------------------------------
// Path encoder: one block (512 threads, 16 waves) per path; both layers fused
// ---------------------------------------------------------------------------
struct EncParams {
  const float* paths;
  const unsigned char* masks;
  const _Float16* qkv_w16[2];
  const _Float16* out_w16[2];
  const _Float16* ff1_w16[2];
  const _Float16* ff2_w16[2];
  const float* ln1_w[2]; const float* ln1_b[2];
  const float* ln2_w[2]; const float* ln2_b[2];
  const float* qkv_b[2]; const float* out_b[2];
  const float* ff1_b[2]; const float* ff2_b[2];
  const float* enc_norm_w; const float* enc_norm_b;
  float* path_emb;       // [NPATH, DMODEL] f32
  _Float16* path_emb16;  // [NPATH, DMODEL] f16
};

__device__ __forceinline__ void ln_rows(const float* H, _Float16* X, const float* w,
                                        const float* b, int waveId, int lane) {
  // 16 waves x 4 rows = 64 rows; wave-local shuffle reduction (wave32)
#pragma unroll
  for (int i = 0; i < 4; ++i) {
    int r = waveId + (i << 4);
    const float* row = H + (size_t)r * LDH;
    float s = 0.f, s2 = 0.f;
    for (int c = lane; c < DMODEL; c += 32) { float x = row[c]; s += x; s2 += x * x; }
#pragma unroll
    for (int off = 16; off > 0; off >>= 1) {
      s += __shfl_xor(s, off); s2 += __shfl_xor(s2, off);
    }
    float m = s * (1.f / DMODEL);
    float var = s2 * (1.f / DMODEL) - m * m;
    float inv = rsqrtf(var + 1e-5f);
    for (int c = lane; c < DMODEL; c += 32)
      X[(size_t)r * LDX + c] = (_Float16)((row[c] - m) * inv * w[c] + b[c]);
  }
}

__global__ __launch_bounds__(512) void path_encoder_kernel(EncParams P) {
  const int p = blockIdx.x;
  const int tid = threadIdx.x;
  const int lane = tid & 31;
  const int waveId = tid >> 5;

  extern __shared__ char smem[];
  float*    h   = (float*)(smem);                 // [64][516] f32 : 132096 B
  _Float16* xln = (_Float16*)(smem + 132096);     // [64][520] f16 :  66560 B
  _Float16* scr = (_Float16*)(smem + 198656);     // [64][520] f16 :  66560 B
  _Float16* qs  = (_Float16*)(smem + 265216);     // [64][72]  f16 :   9216 B
  _Float16* ks  = (_Float16*)(smem + 274432);     // [64][72]  f16 :   9216 B
  _Float16* vT  = (_Float16*)(smem + 283648);     // [64][72]  f16 :   9216 B (V^T)
  float*    sc  = (float*)(smem + 292864);        // [64][68]  f32 :  17408 B
  _Float16* pr  = (_Float16*)(smem + 310272);     // [64][72]  f16 :   9216 B
  // total 319488 B <= 320KB WGP LDS

  // ---- h = paths[p] + positional encoding ----
  const float* pp = P.paths + (size_t)p * NTOK * DMODEL;
  for (int idx = tid; idx < NTOK * DMODEL; idx += 512) {
    int t = idx >> 9, d = idx & 511;
    int i2 = d & ~1;
    float div = __expf(-(float)i2 * (9.2103403719761836f / (float)DMODEL));
    float ang = (float)t * div;
    float pe = (d & 1) ? __cosf(ang) : __sinf(ang);
    h[(size_t)t * LDH + d] = pp[idx] + pe;
  }
  __syncthreads();

  const unsigned char* mrow = P.masks + (size_t)p * NTOK;

  for (int l = 0; l < 2; ++l) {
    // ---------------- attention ----------------
    ln_rows(h, xln, P.ln1_w[l], P.ln1_b[l], waveId, lane);
    __syncthreads();

    for (int hh = 0; hh < NHEAD; ++hh) {
      // Q,K,V tiles at the same (mt,nt) per wave: A fragment shared by 3 WMMAs
      {
        int mt = waveId >> 2, nt = waveId & 3;
        const _Float16* Wq = P.qkv_w16[l] + (size_t)(hh * HDIM) * DMODEL;
        const _Float16* Wk = P.qkv_w16[l] + (size_t)(DMODEL + hh * HDIM) * DMODEL;
        const _Float16* Wv = P.qkv_w16[l] + (size_t)(2 * DMODEL + hh * HDIM) * DMODEL;
        v8f aq = zero8(), ak = zero8(), av = zero8();
        for (int k0 = 0; k0 < DMODEL; k0 += 32) {
          v16h a = ldA(xln, LDX, mt * 16, k0, lane);
          aq = wmma16(a, ldBT(Wq, DMODEL, k0, nt * 16, lane), aq);
          ak = wmma16(a, ldBT(Wk, DMODEL, k0, nt * 16, lane), ak);
          av = wmma16(a, ldBT(Wv, DMODEL, k0, nt * 16, lane), av);
        }
        int nn = hh * HDIM + nt * 16 + (lane & 15);
        float bq = P.qkv_b[l][nn];
        float bk = P.qkv_b[l][DMODEL + nn];
        float bv = P.qkv_b[l][2 * DMODEL + nn];
#pragma unroll
        for (int r8 = 0; r8 < 8; ++r8) { aq[r8] += bq; ak[r8] += bk; av[r8] += bv; }
        stC16(qs, LDQ, mt * 16, nt * 16, lane, aq, false);
        stC16(ks, LDQ, mt * 16, nt * 16, lane, ak, false);
        stC16T(vT, LDQ, mt * 16, nt * 16, lane, av);   // store V transposed
      }
      __syncthreads();

      // scores = q @ k^T * 1/8 : 16 tiles -> 1 tile/wave
      {
        int mt = waveId >> 2, nt = waveId & 3;
        v8f c = zero8();
        for (int k0 = 0; k0 < HDIM; k0 += 32) {
          v16h a = ldA(qs, LDQ, mt * 16, k0, lane);
          v16h b = ldBT(ks, LDQ, k0, nt * 16, lane);   // B[e][n] = k[n][e]
          c = wmma16(a, b, c);
        }
#pragma unroll
        for (int r8 = 0; r8 < 8; ++r8) c[r8] *= 0.125f;
        stC(sc, LDSC, mt * 16, nt * 16, lane, c);
      }
      __syncthreads();

      // masked softmax along keys: wave-per-4-rows, lane-parallel
#pragma unroll
      for (int i = 0; i < 4; ++i) {
        int r = waveId + (i << 4);
        int j0 = lane, j1 = lane + 32;
        bool v0 = mrow[j0] != 0, v1 = mrow[j1] != 0;
        float s0 = sc[(size_t)r * LDSC + j0];
        float s1 = sc[(size_t)r * LDSC + j1];
        float mx = fmaxf(v0 ? s0 : -3.0e38f, v1 ? s1 : -3.0e38f);
#pragma unroll
        for (int off = 16; off > 0; off >>= 1) mx = fmaxf(mx, __shfl_xor(mx, off));
        float e0 = v0 ? __expf(s0 - mx) : 0.f;
        float e1 = v1 ? __expf(s1 - mx) : 0.f;
        float sum = e0 + e1;
#pragma unroll
        for (int off = 16; off > 0; off >>= 1) sum += __shfl_xor(sum, off);
        float inv = 1.f / sum;
        pr[(size_t)r * LDQ + j0] = (_Float16)(e0 * inv);
        pr[(size_t)r * LDQ + j1] = (_Float16)(e1 * inv);
      }
      __syncthreads();

      // o_h = prob @ v -> concat column block hh*64 of scr  (B from V^T, contiguous)
      {
        int mt = waveId >> 2, nt = waveId & 3;
        v8f c = zero8();
        for (int k0 = 0; k0 < HDIM; k0 += 32) {
          v16h a = ldA(pr, LDQ, mt * 16, k0, lane);
          v16h b = ldBT(vT, LDQ, k0, nt * 16, lane);   // B[e][n] = vT[n][e] = v[e][n]
          c = wmma16(a, b, c);
        }
        stC16(scr, LDX, mt * 16, hh * HDIM + nt * 16, lane, c, false);
      }
      __syncthreads();
    }

    // h += scr @ out_w^T + out_b : wave = 2 col-tiles x 4 row-tiles, B reuse x4
    {
      int nt0 = waveId * 2;
      v8f acc[8];
#pragma unroll
      for (int mt = 0; mt < 4; ++mt) {
        acc[mt]     = ldC(h, LDH, mt * 16, nt0 * 16, lane);
        acc[4 + mt] = ldC(h, LDH, mt * 16, (nt0 + 1) * 16, lane);
      }
      for (int k0 = 0; k0 < DMODEL; k0 += 32) {
        v16h b0 = ldBT(P.out_w16[l], DMODEL, k0, nt0 * 16, lane);
        v16h b1 = ldBT(P.out_w16[l], DMODEL, k0, (nt0 + 1) * 16, lane);
#pragma unroll
        for (int mt = 0; mt < 4; ++mt) {
          v16h a = ldA(scr, LDX, mt * 16, k0, lane);
          acc[mt]     = wmma16(a, b0, acc[mt]);
          acc[4 + mt] = wmma16(a, b1, acc[4 + mt]);
        }
      }
      float bias0 = P.out_b[l][nt0 * 16 + (lane & 15)];
      float bias1 = P.out_b[l][(nt0 + 1) * 16 + (lane & 15)];
#pragma unroll
      for (int mt = 0; mt < 4; ++mt) {
#pragma unroll
        for (int r8 = 0; r8 < 8; ++r8) { acc[mt][r8] += bias0; acc[4 + mt][r8] += bias1; }
        stC(h, LDH, mt * 16, nt0 * 16, lane, acc[mt]);
        stC(h, LDH, mt * 16, (nt0 + 1) * 16, lane, acc[4 + mt]);
      }
    }
    __syncthreads();

    // ---------------- feed-forward ----------------
    ln_rows(h, xln, P.ln2_w[l], P.ln2_b[l], waveId, lane);
    __syncthreads();

    for (int c4 = 0; c4 < 4; ++c4) {
      // a_c = relu(xln @ ff1_w[c4*512 : ...]^T + b1)
      {
        int nt0 = waveId * 2;
        const _Float16* W = P.ff1_w16[l] + (size_t)(c4 * DMODEL) * DMODEL;
        v8f acc[8];
#pragma unroll
        for (int t = 0; t < 8; ++t) acc[t] = zero8();
        for (int k0 = 0; k0 < DMODEL; k0 += 32) {
          v16h b0 = ldBT(W, DMODEL, k0, nt0 * 16, lane);
          v16h b1 = ldBT(W, DMODEL, k0, (nt0 + 1) * 16, lane);
#pragma unroll
          for (int mt = 0; mt < 4; ++mt) {
            v16h a = ldA(xln, LDX, mt * 16, k0, lane);
            acc[mt]     = wmma16(a, b0, acc[mt]);
            acc[4 + mt] = wmma16(a, b1, acc[4 + mt]);
          }
        }
        float bias0 = P.ff1_b[l][c4 * DMODEL + nt0 * 16 + (lane & 15)];
        float bias1 = P.ff1_b[l][c4 * DMODEL + (nt0 + 1) * 16 + (lane & 15)];
#pragma unroll
        for (int mt = 0; mt < 4; ++mt) {
#pragma unroll
          for (int r8 = 0; r8 < 8; ++r8) { acc[mt][r8] += bias0; acc[4 + mt][r8] += bias1; }
          stC16(scr, LDX, mt * 16, nt0 * 16, lane, acc[mt], true);
          stC16(scr, LDX, mt * 16, (nt0 + 1) * 16, lane, acc[4 + mt], true);
        }
      }
      __syncthreads();

      // h += a_c @ ff2_w[:, c4*512:...]^T   (+ b2 on last chunk)
      {
        int nt0 = waveId * 2;
        const _Float16* W = P.ff2_w16[l] + c4 * DMODEL;   // row stride FFDIM
        v8f acc[8];
#pragma unroll
        for (int mt = 0; mt < 4; ++mt) {
          acc[mt]     = ldC(h, LDH, mt * 16, nt0 * 16, lane);
          acc[4 + mt] = ldC(h, LDH, mt * 16, (nt0 + 1) * 16, lane);
        }
        for (int k0 = 0; k0 < DMODEL; k0 += 32) {
          v16h b0 = ldBT(W, FFDIM, k0, nt0 * 16, lane);
          v16h b1 = ldBT(W, FFDIM, k0, (nt0 + 1) * 16, lane);
#pragma unroll
          for (int mt = 0; mt < 4; ++mt) {
            v16h a = ldA(scr, LDX, mt * 16, k0, lane);
            acc[mt]     = wmma16(a, b0, acc[mt]);
            acc[4 + mt] = wmma16(a, b1, acc[4 + mt]);
          }
        }
        if (c4 == 3) {
          float bias0 = P.ff2_b[l][nt0 * 16 + (lane & 15)];
          float bias1 = P.ff2_b[l][(nt0 + 1) * 16 + (lane & 15)];
#pragma unroll
          for (int mt = 0; mt < 4; ++mt) {
#pragma unroll
            for (int r8 = 0; r8 < 8; ++r8) { acc[mt][r8] += bias0; acc[4 + mt][r8] += bias1; }
          }
        }
#pragma unroll
        for (int mt = 0; mt < 4; ++mt) {
          stC(h, LDH, mt * 16, nt0 * 16, lane, acc[mt]);
          stC(h, LDH, mt * 16, (nt0 + 1) * 16, lane, acc[4 + mt]);
        }
      }
      __syncthreads();
    }
  }

  // final LN of CLS row -> path_emb
  if (waveId == 0) {
    float s = 0.f, s2 = 0.f;
    for (int c = lane; c < DMODEL; c += 32) { float x = h[c]; s += x; s2 += x * x; }
#pragma unroll
    for (int off = 16; off > 0; off >>= 1) {
      s += __shfl_xor(s, off); s2 += __shfl_xor(s2, off);
    }
    float m = s * (1.f / DMODEL);
    float var = s2 * (1.f / DMODEL) - m * m;
    float inv = rsqrtf(var + 1e-5f);
    for (int c = lane; c < DMODEL; c += 32) {
      float y = (h[c] - m) * inv * P.enc_norm_w[c] + P.enc_norm_b[c];
      P.path_emb[(size_t)p * DMODEL + c] = y;
      P.path_emb16[(size_t)p * DMODEL + c] = (_Float16)y;
    }
  }
}

// ---------------------------------------------------------------------------
// block reductions (256 threads = 8 waves)
// ---------------------------------------------------------------------------
__device__ __forceinline__ float blk_sum(float v, float* red) {
  int lane = threadIdx.x & 31, wid = threadIdx.x >> 5;
#pragma unroll
  for (int off = 16; off > 0; off >>= 1) v += __shfl_xor(v, off);
  __syncthreads();
  if (lane == 0) red[wid] = v;
  __syncthreads();
  float t = 0.f;
#pragma unroll
  for (int i = 0; i < 8; ++i) t += red[i];
  return t;
}
__device__ __forceinline__ float blk_max(float v, float* red) {
  int lane = threadIdx.x & 31, wid = threadIdx.x >> 5;
#pragma unroll
  for (int off = 16; off > 0; off >>= 1) v = fmaxf(v, __shfl_xor(v, off));
  __syncthreads();
  if (lane == 0) red[wid] = v;
  __syncthreads();
  float t = -3.0e38f;
#pragma unroll
  for (int i = 0; i < 8; ++i) t = fmaxf(t, red[i]);
  return t;
}
// LayerNorm over shared array x[n] (optionally relu), 256 threads
__device__ __forceinline__ void ln_shared(float* x, int n, const float* w,
                                          const float* b, float* red, bool dorelu) {
  float s = 0.f, s2 = 0.f;
  for (int j = threadIdx.x; j < n; j += 256) { float v = x[j]; s += v; s2 += v * v; }
  s = blk_sum(s, red);
  s2 = blk_sum(s2, red);
  float m = s / n, var = s2 / n - m * m;
  float inv = rsqrtf(var + 1e-5f);
  for (int j = threadIdx.x; j < n; j += 256) {
    float y = (x[j] - m) * inv * w[j] + b[j];
    x[j] = dorelu ? fmaxf(y, 0.f) : y;
  }
  __syncthreads();
}

// ---------------------------------------------------------------------------
// variant projection: v = relu(LN(variants @ vp_w^T + vp_b))
// ---------------------------------------------------------------------------
__global__ __launch_bounds__(256) void vproj_kernel(const float* __restrict__ variants,
                                                    const float* __restrict__ vp_w,
                                                    const float* __restrict__ vp_b,
                                                    const float* __restrict__ vp_ln_w,
                                                    const float* __restrict__ vp_ln_b,
                                                    float* __restrict__ vbuf) {
  int b = blockIdx.x;
  __shared__ float y[DMODEL];
  __shared__ float red[8];
  const float* xb = variants + (size_t)b * DMODEL;
  for (int j = threadIdx.x; j < DMODEL; j += 256) {
    float s = vp_b[j];
    const float* w = vp_w + (size_t)j * DMODEL;
    for (int k = 0; k < DMODEL; ++k) s += xb[k] * w[k];
    y[j] = s;
  }
  __syncthreads();
  ln_shared(y, DMODEL, vp_ln_w, vp_ln_b, red, true);
  for (int j = threadIdx.x; j < DMODEL; j += 256) vbuf[(size_t)b * DMODEL + j] = y[j];
}

// ---------------------------------------------------------------------------
// k,val = path_emb @ {wk,wv}^T + {bk,bv}  (WMMA; one 16-row slab per block)
// ---------------------------------------------------------------------------
__global__ __launch_bounds__(512) void kv_kernel(const _Float16* __restrict__ pemb16,
                                                 const _Float16* __restrict__ aggw16,
                                                 const float* __restrict__ aggb,
                                                 float* __restrict__ kbuf,
                                                 float* __restrict__ valbuf) {
  int slab = blockIdx.x;             // NPATH/16 = 64
  int lane = threadIdx.x & 31, waveId = threadIdx.x >> 5;
  // wave: 4 consecutive tiles, same matrix -> A fragment shared by 4 WMMAs
  int idx0 = waveId * 4;
  int which = idx0 >> 5;             // 0 = k, 1 = val
  int ntb = idx0 & 31;
  const _Float16* W = aggw16 + (size_t)(DMODEL + which * DMODEL) * DMODEL;
  v8f acc[4];
#pragma unroll
  for (int t = 0; t < 4; ++t) acc[t] = zero8();
  for (int k0 = 0; k0 < DMODEL; k0 += 32) {
    v16h a = ldA(pemb16, DMODEL, slab * 16, k0, lane);
#pragma unroll
    for (int t = 0; t < 4; ++t) {
      v16h b = ldBT(W, DMODEL, k0, (ntb + t) * 16, lane);
      acc[t] = wmma16(a, b, acc[t]);
    }
  }
  float* dst = which ? valbuf : kbuf;
#pragma unroll
  for (int t = 0; t < 4; ++t) {
    float bias = aggb[DMODEL + which * DMODEL + (ntb + t) * 16 + (lane & 15)];
#pragma unroll
    for (int r8 = 0; r8 < 8; ++r8) acc[t][r8] += bias;
    stC(dst, DMODEL, slab * 16, (ntb + t) * 16, lane, acc[t]);
  }
}

// ---------------------------------------------------------------------------
// ragged per-sample attention pooling + features
// ---------------------------------------------------------------------------
__global__ __launch_bounds__(256) void agg_kernel(
    const float* __restrict__ vbuf, const float* __restrict__ kbuf,
    const float* __restrict__ valbuf, const int* __restrict__ pts,
    const float* __restrict__ agg_qkv_w, const float* __restrict__ agg_qkv_b,
    const float* __restrict__ agg_out_w, const float* __restrict__ agg_out_b,
    const float* __restrict__ agg_norm_w, const float* __restrict__ agg_norm_b,
    float* __restrict__ features) {
  int b = blockIdx.x, tid = threadIdx.x;
  __shared__ float q[DMODEL], prob[NPATH], aggv[DMODEL], outv[DMODEL], red[8];
  __shared__ int hasp;
  const float* vb = vbuf + (size_t)b * DMODEL;

  if (tid == 0) hasp = 0;
  for (int j = tid; j < DMODEL; j += 256) {
    float s = agg_qkv_b[j];
    const float* w = agg_qkv_w + (size_t)j * DMODEL;
    for (int k = 0; k < DMODEL; ++k) s += vb[k] * w[k];
    q[j] = s;
  }
  __syncthreads();
  for (int p = tid; p < NPATH; p += 256)
    if (pts[p] == b) atomicOr(&hasp, 1);
  __syncthreads();

  for (int hh = 0; hh < NHEAD; ++hh) {
    for (int p = tid; p < NPATH; p += 256) {
      float s = -3.0e38f;
      if (pts[p] == b) {
        s = 0.f;
        const float* kp = kbuf + (size_t)p * DMODEL + hh * HDIM;
        const float* qh = q + hh * HDIM;
        for (int e = 0; e < HDIM; ++e) s += qh[e] * kp[e];
        s *= 0.125f;
      }
      prob[p] = s;
    }
    float mx = -3.0e38f;
    for (int p = tid; p < NPATH; p += 256) mx = fmaxf(mx, prob[p]);
    mx = blk_max(mx, red);
    float sum = 0.f;
    for (int p = tid; p < NPATH; p += 256) {
      float e = (prob[p] > -1.0e37f) ? __expf(prob[p] - mx) : 0.f;
      prob[p] = e; sum += e;
    }
    sum = blk_sum(sum, red);
    float inv = (sum > 0.f) ? 1.f / sum : 0.f;
    if (tid < HDIM) {
      float s = 0.f;
      for (int p = 0; p < NPATH; ++p) {
        float w = prob[p];
        if (w != 0.f) s += w * valbuf[(size_t)p * DMODEL + hh * HDIM + tid];
      }
      aggv[hh * HDIM + tid] = s * inv;
    }
    __syncthreads();
  }

  for (int j = tid; j < DMODEL; j += 256) {
    float s = agg_out_b[j];
    const float* w = agg_out_w + (size_t)j * DMODEL;
    for (int k = 0; k < DMODEL; ++k) s += aggv[k] * w[k];
    outv[j] = s + vb[j];   // agg + v
  }
  __syncthreads();
  ln_shared(outv, DMODEL, agg_norm_w, agg_norm_b, red, false);
  float* F = features + (size_t)b * DMODEL;
  if (hasp) { for (int j = tid; j < DMODEL; j += 256) F[j] = outv[j]; }
  else      { for (int j = tid; j < DMODEL; j += 256) F[j] = vb[j];  }
}

// ---------------------------------------------------------------------------
// category / gate / label heads
// ---------------------------------------------------------------------------
__global__ __launch_bounds__(256) void heads_kernel(
    const float* __restrict__ features,
    const float* __restrict__ cat_w1, const float* __restrict__ cat_b1,
    const float* __restrict__ cat_ln_w, const float* __restrict__ cat_ln_b,
    const float* __restrict__ cat_w2, const float* __restrict__ cat_b2,
    const float* __restrict__ cat_emb,
    const float* __restrict__ gate_w1, const float* __restrict__ gate_b1,
    const float* __restrict__ gate_w2, const float* __restrict__ gate_b2,
    const float* __restrict__ lbl_w1, const float* __restrict__ lbl_b1,
    const float* __restrict__ lbl_ln_w, const float* __restrict__ lbl_ln_b,
    const float* __restrict__ lbl_w2, const float* __restrict__ lbl_b2,
    const float* __restrict__ lbl_w3, const float* __restrict__ lbl_b3,
    float* __restrict__ label_logits, float* __restrict__ cat_logits) {
  int b = blockIdx.x, tid = threadIdx.x;
  __shared__ float f[DMODEL], ch[DMODEL], cp[NCAT], gh[DMODEL];
  __shared__ float comb[2 * DMODEL], l1[DMODEL], l2[DMODEL / 2], red[8];

  for (int j = tid; j < DMODEL; j += 256) f[j] = features[(size_t)b * DMODEL + j];
  __syncthreads();

  for (int j = tid; j < DMODEL; j += 256) {
    float s = cat_b1[j];
    const float* w = cat_w1 + (size_t)j * DMODEL;
    for (int k = 0; k < DMODEL; ++k) s += f[k] * w[k];
    ch[j] = s;
  }
  __syncthreads();
  ln_shared(ch, DMODEL, cat_ln_w, cat_ln_b, red, true);

  for (int n = tid; n < NCAT; n += 256) {
    float s = cat_b2[n];
    const float* w = cat_w2 + (size_t)n * DMODEL;
    for (int k = 0; k < DMODEL; ++k) s += ch[k] * w[k];
    cat_logits[(size_t)b * NCAT + n] = s;
    cp[n] = 1.f / (1.f + __expf(-s));
  }
  __syncthreads();

  for (int j = tid; j < DMODEL; j += 256) {
    float s = gate_b1[j];
    const float* w = gate_w1 + (size_t)j * NCAT;
    for (int n = 0; n < NCAT; ++n) s += cp[n] * w[n];
    gh[j] = fmaxf(s, 0.f);
  }
  __syncthreads();

  for (int j = tid; j < DMODEL; j += 256) {
    float s = gate_b2[j];
    const float* w = gate_w2 + (size_t)j * DMODEL;
    for (int k = 0; k < DMODEL; ++k) s += gh[k] * w[k];
    float g = 1.f / (1.f + __expf(-s));
    float wce = 0.f;
    for (int n = 0; n < NCAT; ++n) wce += cp[n] * cat_emb[(size_t)n * DMODEL + j];
    comb[j] = f[j];
    comb[DMODEL + j] = f[j] + g * wce;
  }
  __syncthreads();

  for (int j = tid; j < DMODEL; j += 256) {
    float s = lbl_b1[j];
    const float* w = lbl_w1 + (size_t)j * (2 * DMODEL);
    for (int k = 0; k < 2 * DMODEL; ++k) s += comb[k] * w[k];
    l1[j] = s;
  }
  __syncthreads();
  ln_shared(l1, DMODEL, lbl_ln_w, lbl_ln_b, red, true);

  for (int j = tid; j < DMODEL / 2; j += 256) {
    float s = lbl_b2[j];
    const float* w = lbl_w2 + (size_t)j * DMODEL;
    for (int k = 0; k < DMODEL; ++k) s += l1[k] * w[k];
    l2[j] = fmaxf(s, 0.f);
  }
  __syncthreads();

  for (int n = tid; n < NCLS; n += 256) {
    float s = lbl_b3[n];
    const float* w = lbl_w3 + (size_t)n * (DMODEL / 2);
    for (int k = 0; k < DMODEL / 2; ++k) s += l2[k] * w[k];
    label_logits[(size_t)b * NCLS + n] = s;
  }
}

// ---------------------------------------------------------------------------
// launcher
// ---------------------------------------------------------------------------
extern "C" void kernel_launch(void* const* d_in, const int* in_sizes, int n_in,
                              void* d_out, int out_size, void* d_ws, size_t ws_size,
                              hipStream_t stream) {
  (void)in_sizes; (void)n_in; (void)out_size; (void)ws_size;

  // Input order: DFS over sorted dict keys (jax tree order), top-level dict order.
  const float* agg_norm_b = (const float*)d_in[0];
  const float* agg_norm_w = (const float*)d_in[1];
  const float* agg_out_b  = (const float*)d_in[2];
  const float* agg_out_w  = (const float*)d_in[3];
  const float* agg_qkv_b  = (const float*)d_in[4];
  const float* agg_qkv_w  = (const float*)d_in[5];
  const float* cat_b1     = (const float*)d_in[6];
  const float* cat_b2     = (const float*)d_in[7];
  const float* cat_emb    = (const float*)d_in[8];
  const float* cat_ln_b   = (const float*)d_in[9];
  const float* cat_ln_w   = (const float*)d_in[10];
  const float* cat_w1     = (const float*)d_in[11];
  const float* cat_w2     = (const float*)d_in[12];
  const float* enc_norm_b = (const float*)d_in[13];
  const float* enc_norm_w = (const float*)d_in[14];
  const float* gate_b1    = (const float*)d_in[15];
  const float* gate_b2    = (const float*)d_in[16];
  const float* gate_w1    = (const float*)d_in[17];
  const float* gate_w2    = (const float*)d_in[18];
  const float *Lff1_b[2], *Lff1_w[2], *Lff2_b[2], *Lff2_w[2];
  const float *Lln1_b[2], *Lln1_w[2], *Lln2_b[2], *Lln2_w[2];
  const float *Lout_b[2], *Lout_w[2], *Lqkv_b[2], *Lqkv_w[2];
  for (int l = 0; l < 2; ++l) {
    int base = 19 + 12 * l;
    Lff1_b[l] = (const float*)d_in[base + 0];
    Lff1_w[l] = (const float*)d_in[base + 1];
    Lff2_b[l] = (const float*)d_in[base + 2];
    Lff2_w[l] = (const float*)d_in[base + 3];
    Lln1_b[l] = (const float*)d_in[base + 4];
    Lln1_w[l] = (const float*)d_in[base + 5];
    Lln2_b[l] = (const float*)d_in[base + 6];
    Lln2_w[l] = (const float*)d_in[base + 7];
    Lout_b[l] = (const float*)d_in[base + 8];
    Lout_w[l] = (const float*)d_in[base + 9];
    Lqkv_b[l] = (const float*)d_in[base + 10];
    Lqkv_w[l] = (const float*)d_in[base + 11];
  }
  const float* lbl_b1   = (const float*)d_in[43];
  const float* lbl_b2   = (const float*)d_in[44];
  const float* lbl_b3   = (const float*)d_in[45];
  const float* lbl_ln_b = (const float*)d_in[46];
  const float* lbl_ln_w = (const float*)d_in[47];
  const float* lbl_w1   = (const float*)d_in[48];
  const float* lbl_w2   = (const float*)d_in[49];
  const float* lbl_w3   = (const float*)d_in[50];
  const float* vp_b     = (const float*)d_in[51];
  const float* vp_ln_b  = (const float*)d_in[52];
  const float* vp_ln_w  = (const float*)d_in[53];
  const float* vp_w     = (const float*)d_in[54];
  const float* variants = (const float*)d_in[55];
  const float* paths    = (const float*)d_in[56];
  const unsigned char* masks = (const unsigned char*)d_in[57];
  const int* pts        = (const int*)d_in[58];

  // ---- workspace layout ----
  char* ws = (char*)d_ws;
  size_t off = 0;
  auto take = [&](size_t bytes) -> void* {
    void* p = ws + off;
    off += (bytes + 255) & ~(size_t)255;
    return p;
  };
  _Float16* qkv16[2]; _Float16* out16[2]; _Float16* ff1_16[2]; _Float16* ff2_16[2];
  for (int l = 0; l < 2; ++l) {
    qkv16[l]  = (_Float16*)take((size_t)3 * DMODEL * DMODEL * 2);
    out16[l]  = (_Float16*)take((size_t)DMODEL * DMODEL * 2);
    ff1_16[l] = (_Float16*)take((size_t)FFDIM * DMODEL * 2);
    ff2_16[l] = (_Float16*)take((size_t)DMODEL * FFDIM * 2);
  }
  _Float16* agg16   = (_Float16*)take((size_t)3 * DMODEL * DMODEL * 2);
  float*    pemb    = (float*)take((size_t)NPATH * DMODEL * 4);
  _Float16* pemb16  = (_Float16*)take((size_t)NPATH * DMODEL * 2);
  float*    kbuf    = (float*)take((size_t)NPATH * DMODEL * 4);
  float*    valbuf  = (float*)take((size_t)NPATH * DMODEL * 4);
  float*    vbuf    = (float*)take((size_t)NSAMP * DMODEL * 4);

  auto cvt = [&](const float* src, _Float16* dst, long n) {
    int grid = (int)((n + 1023) / 1024);
    if (grid > 2048) grid = 2048;
    f32_to_f16_kernel<<<grid, 256, 0, stream>>>(src, dst, n);
  };
  for (int l = 0; l < 2; ++l) {
    cvt(Lqkv_w[l], qkv16[l],  (long)3 * DMODEL * DMODEL);
    cvt(Lout_w[l], out16[l],  (long)DMODEL * DMODEL);
    cvt(Lff1_w[l], ff1_16[l], (long)FFDIM * DMODEL);
    cvt(Lff2_w[l], ff2_16[l], (long)DMODEL * FFDIM);
  }
  cvt(agg_qkv_w, agg16, (long)3 * DMODEL * DMODEL);

  // ---- path encoder ----
  EncParams EP;
  EP.paths = paths; EP.masks = masks;
  for (int l = 0; l < 2; ++l) {
    EP.qkv_w16[l] = qkv16[l]; EP.out_w16[l] = out16[l];
    EP.ff1_w16[l] = ff1_16[l]; EP.ff2_w16[l] = ff2_16[l];
    EP.ln1_w[l] = Lln1_w[l]; EP.ln1_b[l] = Lln1_b[l];
    EP.ln2_w[l] = Lln2_w[l]; EP.ln2_b[l] = Lln2_b[l];
    EP.qkv_b[l] = Lqkv_b[l]; EP.out_b[l] = Lout_b[l];
    EP.ff1_b[l] = Lff1_b[l]; EP.ff2_b[l] = Lff2_b[l];
  }
  EP.enc_norm_w = enc_norm_w; EP.enc_norm_b = enc_norm_b;
  EP.path_emb = pemb; EP.path_emb16 = pemb16;
  path_encoder_kernel<<<NPATH, 512, 319488, stream>>>(EP);

  // ---- variant projection ----
  vproj_kernel<<<NSAMP, 256, 0, stream>>>(variants, vp_w, vp_b, vp_ln_w, vp_ln_b, vbuf);

  // ---- k / val projections (WMMA) ----
  kv_kernel<<<NPATH / 16, 512, 0, stream>>>(pemb16, agg16, agg_qkv_b, kbuf, valbuf);

  // ---- output layout: [label_logits | cat_logits | features] ----
  float* out = (float*)d_out;
  float* o_label = out;
  float* o_cat   = out + (size_t)NSAMP * NCLS;
  float* o_feat  = out + (size_t)NSAMP * NCLS + (size_t)NSAMP * NCAT;

  // ---- ragged attention pooling + features ----
  agg_kernel<<<NSAMP, 256, 0, stream>>>(vbuf, kbuf, valbuf, pts,
                                        agg_qkv_w, agg_qkv_b, agg_out_w, agg_out_b,
                                        agg_norm_w, agg_norm_b, o_feat);

  // ---- heads ----
  heads_kernel<<<NSAMP, 256, 0, stream>>>(o_feat,
                                          cat_w1, cat_b1, cat_ln_w, cat_ln_b,
                                          cat_w2, cat_b2, cat_emb,
                                          gate_w1, gate_b1, gate_w2, gate_b2,
                                          lbl_w1, lbl_b1, lbl_ln_w, lbl_ln_b,
                                          lbl_w2, lbl_b2, lbl_w3, lbl_b3,
                                          o_label, o_cat);
}
